// CleanLSTM_7653631722037
// MI455X (gfx1250) — compile-verified, matrix operand order checked by reading, not run on hardware
//
#include <hip/hip_runtime.h>
#include <hip/hip_bf16.h>

typedef __bf16 v16bf __attribute__((ext_vector_type(16)));
typedef __bf16 v4bf  __attribute__((ext_vector_type(4)));
typedef float  v8f   __attribute__((ext_vector_type(8)));
typedef float  v4f   __attribute__((ext_vector_type(4)));

struct BFrag { v4f lo, hi; };

#define BM 128
#define BN 256
#define WM 64
#define WN 64

// C[M,N] (f32) = A[M,K] (bf16, row-major, ld=K) x W[N,K]^T (bf16, row-major, ld=K)
// K is a compile-time constant -> fully unrolled K loop, immediate-offset b128
// loads from 8 precomputed base pointers. Wave tile 64x64: 8 fragment loads
// feed 16 WMMAs per K-step (1 MMA per b128 load). Optional fused per-column
// sum/sumsq partials (BatchNorm stats) via LDS ds_add_f32, optional fused bias.
template <int K>
__global__ __launch_bounds__(256)
void gemm_bf16_kernel(float* __restrict__ C,
                      const __bf16* __restrict__ A,
                      const __bf16* __restrict__ W,
                      int M, int N, int ldc,
                      float* __restrict__ partials,   // [M/BM][N][2] or null
                      const float* __restrict__ bias) // [N] or null
{
    __shared__ float s_stat[BN * 2];

    const int tid  = threadIdx.x;
    const int lane = tid & 31;
    const int wave = tid >> 5;        // 0..7
    const int wm   = wave >> 2;       // 0..1  (M direction)
    const int wn   = wave & 3;        // 0..3  (N direction)
    const int bn0  = blockIdx.x * BN;
    const int bm0  = blockIdx.y * BM;

    for (int i = tid; i < BN * 2; i += 256) s_stat[i] = 0.f;
    __syncthreads();

    v8f acc[4][4];
#pragma unroll
    for (int mi = 0; mi < 4; ++mi)
#pragma unroll
        for (int ni = 0; ni < 4; ++ni)
            acc[mi][ni] = (v8f){0.f, 0.f, 0.f, 0.f, 0.f, 0.f, 0.f, 0.f};

    // Per CDNA5 ISA 16-bit operand layout: lanes 0-15 hold K k..k+7 (VGPR0-3)
    // and K k+16..k+23 (VGPR4-7); lanes 16-31 hold k+8..k+15 and k+24..k+31.
    const int arow = bm0 + wm * WM + (lane & 15);
    const int wrow = bn0 + wn * WN + (lane & 15);
    const int koff = (lane >> 4) << 3;   // 0 or 8 elements

    const __bf16* ap[4];
    const __bf16* wp[4];
#pragma unroll
    for (int mi = 0; mi < 4; ++mi)
        ap[mi] = A + (size_t)(arow + mi * 16) * K + koff;
#pragma unroll
    for (int ni = 0; ni < 4; ++ni)
        wp[ni] = W + (size_t)(wrow + ni * 16) * K + koff;

#pragma unroll
    for (int k = 0; k < K; k += 32) {
        v16bf a[4], b[4];
#pragma unroll
        for (int mi = 0; mi < 4; ++mi) {
            BFrag f;
            f.lo = *(const v4f*)(ap[mi] + k);        // immediate offset k*2
            f.hi = *(const v4f*)(ap[mi] + k + 16);   // immediate offset k*2+32
            a[mi] = __builtin_bit_cast(v16bf, f);
        }
#pragma unroll
        for (int ni = 0; ni < 4; ++ni) {
            BFrag f;
            f.lo = *(const v4f*)(wp[ni] + k);
            f.hi = *(const v4f*)(wp[ni] + k + 16);
            b[ni] = __builtin_bit_cast(v16bf, f);
        }
#pragma unroll
        for (int mi = 0; mi < 4; ++mi)
#pragma unroll
            for (int ni = 0; ni < 4; ++ni)
                acc[mi][ni] = __builtin_amdgcn_wmma_f32_16x16x32_bf16(
                    false, a[mi], false, b[ni], (short)0, acc[mi][ni],
                    false, false);
    }

    // Epilogue: C/D layout — lane n (0-15) col = n; VGPR r -> row r (lanes
    // 0-15) or row 8+r (lanes 16-31).
    const int hi8 = (lane >> 4) << 3;
#pragma unroll
    for (int mi = 0; mi < 4; ++mi) {
        const int mrow = bm0 + wm * WM + mi * 16 + hi8;
#pragma unroll
        for (int ni = 0; ni < 4; ++ni) {
            const int col = bn0 + wn * WN + ni * 16 + (lane & 15);
            const float bval = bias ? bias[col] : 0.f;
            v8f v = acc[mi][ni];
            float s = 0.f, q = 0.f;
#pragma unroll
            for (int r = 0; r < 8; ++r) {
                const float e = v[r];
                C[(size_t)(mrow + r) * ldc + col] = e + bval;
                s += e;
                q += e * e;
            }
            if (partials) {
                const int cl = wn * WN + ni * 16 + (lane & 15);
                atomicAdd(&s_stat[cl * 2 + 0], s);  // ds_add_f32
                atomicAdd(&s_stat[cl * 2 + 1], q);
            }
        }
    }
    if (partials) {
        __syncthreads();
        for (int i = tid; i < BN * 2; i += 256) {
            const int col = bn0 + (i >> 1);
            partials[((size_t)blockIdx.y * N + col) * 2 + (i & 1)] = s_stat[i];
        }
    }
}

// Fold per-row-block partials into BN scale/shift:
// scale = gamma * rsqrt(var + eps), shift = beta - mean * scale
__global__ void bn_finalize_kernel(const float* __restrict__ partials,
                                   const float* __restrict__ gamma,
                                   const float* __restrict__ beta,
                                   float* __restrict__ scale,
                                   float* __restrict__ shift,
                                   int N, int MB, float invB)
{
    const int col = blockIdx.x * blockDim.x + threadIdx.x;
    if (col >= N) return;
    float s = 0.f, q = 0.f;
    for (int mb = 0; mb < MB; ++mb) {
        const float* p = partials + ((size_t)mb * N + col) * 2;
        s += p[0];
        q += p[1];
    }
    const float m   = s * invB;
    const float var = fmaxf(q * invB - m * m, 0.f);
    const float sc  = gamma[col] * rsqrtf(var + 1e-5f);
    scale[col] = sc;
    shift[col] = beta[col] - m * sc;
}

// gates = BN(zx) + BN(zh); chunk order f,i,o,g; LSTM cell update.
// Vectorized x4: pure b128 streaming on the bandwidth-critical pass.
__global__ void lstm_pointwise_kernel(const float* __restrict__ zx,
                                      const float* __restrict__ zh,
                                      const float* __restrict__ scx,
                                      const float* __restrict__ shx,
                                      const float* __restrict__ sch,
                                      const float* __restrict__ shh,
                                      const float* __restrict__ cprev,
                                      float* __restrict__ cout,
                                      float* __restrict__ hout,
                                      __bf16* __restrict__ hbf,
                                      int B, int H)
{
    const int t  = blockIdx.x * blockDim.x + threadIdx.x;
    const int hq = H >> 2;                    // groups of 4 columns
    if (t >= B * hq) return;
    const int b  = t / hq;
    const int hc = (t - b * hq) << 2;
    const size_t gbase = (size_t)b * 4 * H;
    const size_t obase = (size_t)b * H + hc;

    v4f zxv[4], zhv[4], sxv[4], bxv[4], shv[4], bhv[4];
#pragma unroll
    for (int g = 0; g < 4; ++g) {
        const int c = g * H + hc;
        zxv[g] = *(const v4f*)(zx + gbase + c);
        zhv[g] = *(const v4f*)(zh + gbase + c);
        sxv[g] = *(const v4f*)(scx + c);
        bxv[g] = *(const v4f*)(shx + c);
        shv[g] = *(const v4f*)(sch + c);
        bhv[g] = *(const v4f*)(shh + c);
    }
    const v4f cp = *(const v4f*)(cprev + obase);

    v4f cv, hv;
    v4bf hb;
#pragma unroll
    for (int j = 0; j < 4; ++j) {
        const float gf = zxv[0][j] * sxv[0][j] + bxv[0][j] + zhv[0][j] * shv[0][j] + bhv[0][j];
        const float gi = zxv[1][j] * sxv[1][j] + bxv[1][j] + zhv[1][j] * shv[1][j] + bhv[1][j];
        const float go = zxv[2][j] * sxv[2][j] + bxv[2][j] + zhv[2][j] * shv[2][j] + bhv[2][j];
        const float gg = zxv[3][j] * sxv[3][j] + bxv[3][j] + zhv[3][j] * shv[3][j] + bhv[3][j];
        const float sf = 1.f / (1.f + __expf(-gf));
        const float si = 1.f / (1.f + __expf(-gi));
        const float so = 1.f / (1.f + __expf(-go));
        const float c2 = sf * cp[j] + si * tanhf(gg);
        const float h2 = so * tanhf(c2);
        cv[j] = c2;
        hv[j] = h2;
        hb[j] = (__bf16)h2;
    }
    *(v4f*)(cout + obase)  = cv;
    *(v4f*)(hout + obase)  = hv;
    *(v4bf*)(hbf + obase)  = hb;
}

// f32 -> bf16, equal width, vectorized x4 (b128 load, b64 store).
__global__ void cvt_bf16_vec4_kernel(__bf16* __restrict__ dst,
                                     const float* __restrict__ src, int n4)
{
    const int i = blockIdx.x * blockDim.x + threadIdx.x;
    if (i >= n4) return;
    const v4f v = ((const v4f*)src)[i];
    v4bf o;
#pragma unroll
    for (int j = 0; j < 4; ++j) o[j] = (__bf16)v[j];
    ((v4bf*)dst)[i] = o;
}

// f32 [rows, scols] -> bf16 [rows, dcols], zero-padded K tail (Wx0: 257->288).
__global__ void pack_bf16_kernel(__bf16* __restrict__ dst,
                                 const float* __restrict__ src,
                                 int rows, int scols, int dcols)
{
    const int i = blockIdx.x * blockDim.x + threadIdx.x;
    if (i >= rows * dcols) return;
    const int r = i / dcols, c = i % dcols;
    dst[i] = (c < scols) ? (__bf16)src[(size_t)r * scols + c] : (__bf16)0.f;
}

// concat([x, y], axis=1) -> bf16 [B, dcols] zero-padded (257 -> 288).
__global__ void pack_inp_kernel(__bf16* __restrict__ dst,
                                const float* __restrict__ x,
                                const float* __restrict__ y,
                                int B, int XD, int dcols)
{
    const int i = blockIdx.x * blockDim.x + threadIdx.x;
    if (i >= B * dcols) return;
    const int r = i / dcols, c = i % dcols;
    float v = 0.f;
    if (c < XD)       v = x[(size_t)r * XD + c];
    else if (c == XD) v = y[r];
    dst[i] = (__bf16)v;
}

extern "C" void kernel_launch(void* const* d_in, const int* in_sizes, int n_in,
                              void* d_out, int out_size, void* d_ws, size_t ws_size,
                              hipStream_t stream)
{
    const int B = 16384, H = 512, G = 2048, XD = 256, IN0 = 257, KX0 = 288,
              OUT = 256, MB = B / BM;

    const float* x   = (const float*)d_in[0];
    const float* y   = (const float*)d_in[1];
    const float* h0  = (const float*)d_in[2];
    const float* c0  = (const float*)d_in[3];
    const float* Wx0 = (const float*)d_in[4];
    const float* Wh0 = (const float*)d_in[5];
    const float* gx0 = (const float*)d_in[6],  *bx0 = (const float*)d_in[7];
    const float* gh0 = (const float*)d_in[8],  *bh0 = (const float*)d_in[9];
    const float* Wx1 = (const float*)d_in[10], *Wh1 = (const float*)d_in[11];
    const float* gx1 = (const float*)d_in[12], *bx1 = (const float*)d_in[13];
    const float* gh1 = (const float*)d_in[14], *bh1 = (const float*)d_in[15];
    const float* Wo  = (const float*)d_in[16], *bo  = (const float*)d_in[17];

    float* out = (float*)d_out;               // [B, OUT]
    float* h1  = out + (size_t)B * OUT;       // h stack [2,B,H]
    float* h2  = h1 + (size_t)B * H;
    float* c1  = h2 + (size_t)B * H;          // c stack [2,B,H]
    float* c2  = c1 + (size_t)B * H;

    // ---- workspace partition (all offsets 256B aligned) ----
    char* wsb = (char*)d_ws;
    size_t off = 0;
    auto alloc = [&](size_t bytes) -> void* {
        void* p = wsb + off;
        off += (bytes + 255) & ~(size_t)255;
        return p;
    };
    float*  zx    = (float*)alloc((size_t)B * G * 4);       // 134 MB
    float*  zh    = (float*)alloc((size_t)B * G * 4);       // 134 MB
    __bf16* actx  = (__bf16*)alloc((size_t)B * H * 2);      // 16 MB (also holds [B,288])
    __bf16* acth  = (__bf16*)alloc((size_t)B * H * 2);      // 16 MB
    __bf16* wbA   = (__bf16*)alloc((size_t)G * H * 2);      // 2 MB
    __bf16* wbB   = (__bf16*)alloc((size_t)G * H * 2);      // 2 MB
    __bf16* wbO   = (__bf16*)alloc((size_t)OUT * H * 2);
    float*  partX = (float*)alloc((size_t)MB * G * 2 * 4);  // 2 MB
    float*  partH = (float*)alloc((size_t)MB * G * 2 * 4);
    float*  scX   = (float*)alloc(G * 4);
    float*  shX   = (float*)alloc(G * 4);
    float*  scH   = (float*)alloc(G * 4);
    float*  shH   = (float*)alloc(G * 4);
    (void)ws_size; (void)in_sizes; (void)n_in; (void)out_size;

    const dim3 blk(256);
    auto cdiv = [](long long a, long long b) { return (unsigned)((a + b - 1) / b); };
    const dim3 ggrid(G / BN, B / BM);     // (8, 128)
    const dim3 ogrid(OUT >= BN ? OUT / BN : 1, B / BM);  // (1, 128)
    const float invB = 1.f / (float)B;

    // ================= Layer 0 =================
    pack_inp_kernel    <<<cdiv((long long)B * KX0, 256), blk, 0, stream>>>(actx, x, y, B, XD, KX0);
    cvt_bf16_vec4_kernel<<<cdiv((long long)B * H / 4, 256), blk, 0, stream>>>(acth, h0, B * H / 4);
    pack_bf16_kernel   <<<cdiv((long long)G * KX0, 256), blk, 0, stream>>>(wbA, Wx0, G, IN0, KX0);
    cvt_bf16_vec4_kernel<<<cdiv((long long)G * H / 4, 256), blk, 0, stream>>>(wbB, Wh0, G * H / 4);

    gemm_bf16_kernel<288><<<ggrid, blk, 0, stream>>>(zx, actx, wbA, B, G, G, partX, nullptr);
    gemm_bf16_kernel<512><<<ggrid, blk, 0, stream>>>(zh, acth, wbB, B, G, G, partH, nullptr);

    bn_finalize_kernel<<<cdiv(G, 256), blk, 0, stream>>>(partX, gx0, bx0, scX, shX, G, MB, invB);
    bn_finalize_kernel<<<cdiv(G, 256), blk, 0, stream>>>(partH, gh0, bh0, scH, shH, G, MB, invB);

    lstm_pointwise_kernel<<<cdiv((long long)B * H / 4, 256), blk, 0, stream>>>(
        zx, zh, scX, shX, scH, shH, c0, c1, h1, actx, B, H);

    // ================= Layer 1 =================
    cvt_bf16_vec4_kernel<<<cdiv((long long)B * H / 4, 256), blk, 0, stream>>>(acth, h0 + (size_t)B * H, B * H / 4);
    cvt_bf16_vec4_kernel<<<cdiv((long long)G * H / 4, 256), blk, 0, stream>>>(wbA, Wx1, G * H / 4);
    cvt_bf16_vec4_kernel<<<cdiv((long long)G * H / 4, 256), blk, 0, stream>>>(wbB, Wh1, G * H / 4);

    gemm_bf16_kernel<512><<<ggrid, blk, 0, stream>>>(zx, actx, wbA, B, G, G, partX, nullptr);
    gemm_bf16_kernel<512><<<ggrid, blk, 0, stream>>>(zh, acth, wbB, B, G, G, partH, nullptr);

    bn_finalize_kernel<<<cdiv(G, 256), blk, 0, stream>>>(partX, gx1, bx1, scX, shX, G, MB, invB);
    bn_finalize_kernel<<<cdiv(G, 256), blk, 0, stream>>>(partH, gh1, bh1, scH, shH, G, MB, invB);

    lstm_pointwise_kernel<<<cdiv((long long)B * H / 4, 256), blk, 0, stream>>>(
        zx, zh, scX, shX, scH, shH, c0 + (size_t)B * H, c2, h2, actx, B, H);

    // ================= Output projection =================
    cvt_bf16_vec4_kernel<<<cdiv((long long)OUT * H / 4, 256), blk, 0, stream>>>(wbO, Wo, OUT * H / 4);
    gemm_bf16_kernel<512><<<ogrid, blk, 0, stream>>>(out, actx, wbO, B, OUT, OUT, nullptr, bo);
}
